// get_model_78056735638060
// MI455X (gfx1250) — matrix operand order repaired and checked by reading, not compiled, not run on hardware
//
#include <hip/hip_runtime.h>

// ---------------------------------------------------------------------------
// Types for gfx1250 WMMA (wave32): V_WMMA_F32_16X16X32_BF16
// ---------------------------------------------------------------------------
typedef __attribute__((ext_vector_type(16))) __bf16 v16bf;
typedef __attribute__((ext_vector_type(8)))  float  v8f;
typedef __attribute__((ext_vector_type(4)))  unsigned v4u;
typedef __attribute__((ext_vector_type(8)))  unsigned v8u;

union FragU {
    uint4 q[2];
    v16bf v;
};

__device__ __forceinline__ v8f vzero8() {
    v8f z;
#pragma unroll
    for (int i = 0; i < 8; ++i) z[i] = 0.0f;
    return z;
}

__device__ __forceinline__ unsigned short f2bf(float x) {
    unsigned int u = __float_as_uint(x);
    u += 0x7FFFu + ((u >> 16) & 1u);      // round-to-nearest-even
    return (unsigned short)(u >> 16);
}

// ---------------------------------------------------------------------------
// CDNA5 async global->LDS copy (16B per lane), tracked by ASYNCcnt.
// LDS operand is the 32-bit LDS byte offset; for generic pointers to
// __shared__ data the low 32 bits are the LDS offset (flat aperture rule).
// ---------------------------------------------------------------------------
__device__ __forceinline__ void async_cp16(void* lds_dst, const void* g_src) {
    unsigned loff = (unsigned)(unsigned long long)(size_t)lds_dst;
    asm volatile("global_load_async_to_lds_b128 %0, %1, off"
                 :: "v"(loff), "v"(g_src)
                 : "memory");
}
__device__ __forceinline__ void wait_async0() {
    asm volatile("s_wait_asynccnt 0x0" ::: "memory");
}
__device__ __forceinline__ void wait_async4() {
    asm volatile("s_wait_asynccnt 0x4" ::: "memory");
}
__device__ __forceinline__ void wait_tensor0() {
    asm volatile("s_wait_tensorcnt 0x0" ::: "memory");
}
__device__ __forceinline__ void wait_tensor1() {
    asm volatile("s_wait_tensorcnt 0x1" ::: "memory");
}

// ---------------------------------------------------------------------------
// CDNA5 Tensor Data Mover: 2D tile load, global -> LDS (TENSORcnt).
// D# groups per CDNA5 ISA (2-operand form, <=2D).
//   tile_w elems wide (bf16), tile_h rows, row stride stride_elems.
// Issue from one wave only (caller guards); values are wave-uniform -> SGPRs.
// ---------------------------------------------------------------------------
__device__ __forceinline__ void tdm_load_2d(void* lds_dst, const void* g_src,
                                            unsigned tile_w, unsigned tile_h,
                                            unsigned stride_elems) {
    unsigned long long ga = (unsigned long long)(size_t)g_src;
    v4u g0;
    g0.x = 1u;                                             // count=1, user mode
    g0.y = (unsigned)(unsigned long long)(size_t)lds_dst;  // lds_addr (bytes)
    g0.z = (unsigned)ga;                                   // global_addr[31:0]
    g0.w = (unsigned)((ga >> 32) & 0x01FFFFFFull)          // global_addr[56:32]
         | (2u << 30);                                     // type=2 ("image")
    v8u g1;
    g1[0] = 0x00010000u;              // wg_mask=0, data_size=1 (2 bytes)
    g1[1] = 0u;                       // barrier_addr=0, tensor_dim0[15:0]=0 (dim0=1<<20)
    g1[2] = 0x10u;                    // tensor_dim0[31:16]=0x10, tensor_dim1[15:0]=0
    g1[3] = 0x10u | (tile_w << 16);   // tensor_dim1[31:16]=0x10, tile_dim0
    g1[4] = tile_h;                   // tile_dim1, tile_dim2=0
    g1[5] = stride_elems;             // tensor_dim0_stride[31:0]
    g1[6] = 0u;                       // stride hi, tensor_dim1_stride lo
    g1[7] = 0u;
    asm volatile("tensor_load_to_lds %0, %1" :: "s"(g0), "s"(g1) : "memory");
}

// A-matrix fragment (16x32 bf16) from LDS tile [rows][ld], row-major.
// ISA layout: lanes0-15: K=0..7 & 16..23 ; lanes16-31: K=8..15 & 24..31.
__device__ __forceinline__ v16bf lds_frag_a(const unsigned short* t, int row0, int ld, int kk) {
    const int lane = threadIdx.x & 31;
    const int r  = row0 + (lane & 15);
    const int k0 = kk + ((lane >> 4) << 3);
    FragU f;
    f.q[0] = *(const uint4*)(t + (size_t)r * ld + k0);
    f.q[1] = *(const uint4*)(t + (size_t)r * ld + k0 + 16);
    return f.v;
}

// B-matrix fragment (32x16 bf16) from LDS tile holding B^T, i.e. [N][K] row-major.
// ISA layout (B): lanes0-15: K=0..7 & 16..23 ; lanes16-31: K=8..15 & 24..31.
__device__ __forceinline__ v16bf lds_frag_b(const unsigned short* t, int row0, int ld, int kk) {
    const int lane = threadIdx.x & 31;
    const int r  = row0 + (lane & 15);
    const int k0 = kk + ((lane >> 4) << 3);
    const uint4* p = (const uint4*)(t + (size_t)r * ld + k0);
    FragU f;
    f.q[0] = p[0];
    f.q[1] = p[2];
    return f.v;
}

__device__ __forceinline__ v8f wmma_bf16(v16bf a, v16bf b, v8f c) {
    return __builtin_amdgcn_wmma_f32_16x16x32_bf16(false, a, false, b, (short)0, c, false, false);
}

// ---------------------------------------------------------------------------
// Utility kernels
// ---------------------------------------------------------------------------
__global__ void zero_f32_kernel(float* __restrict__ p, size_t n) {
    size_t i = (size_t)blockIdx.x * blockDim.x + threadIdx.x;
    if (i < n) p[i] = 0.0f;
}

__global__ void cvt_bf16_kernel(const float* __restrict__ s, unsigned short* __restrict__ d, size_t n) {
    size_t i = (size_t)blockIdx.x * blockDim.x + threadIdx.x;
    if (i < n) d[i] = f2bf(s[i]);
}

// w [K][N] fp32 -> wT [N][K] bf16
__global__ void wtrans_kernel(const float* __restrict__ w, unsigned short* __restrict__ wt, int K, int N) {
    int i = blockIdx.x * blockDim.x + threadIdx.x;
    if (i < K * N) {
        int k = i / N, n = i - k * N;
        wt[(size_t)n * K + k] = f2bf(w[i]);
    }
}

// ---------------------------------------------------------------------------
// LayerNorm (row length 256): one wave per row, bf16 output
// ---------------------------------------------------------------------------
__global__ __launch_bounds__(256) void layernorm_kernel(
    const float* __restrict__ x, const float* __restrict__ g, const float* __restrict__ bta,
    unsigned short* __restrict__ outH)
{
    const int lane = threadIdx.x & 31;
    const int wave = threadIdx.x >> 5;
    const size_t row = (size_t)blockIdx.x * 8 + wave;
    const float* xr = x + row * 256;

    float vals[8];
    float4 v0 = *(const float4*)&xr[lane * 8];
    float4 v1 = *(const float4*)&xr[lane * 8 + 4];
    vals[0] = v0.x; vals[1] = v0.y; vals[2] = v0.z; vals[3] = v0.w;
    vals[4] = v1.x; vals[5] = v1.y; vals[6] = v1.z; vals[7] = v1.w;

    float s = 0.0f, sq = 0.0f;
#pragma unroll
    for (int j = 0; j < 8; ++j) { s += vals[j]; sq += vals[j] * vals[j]; }
#pragma unroll
    for (int off = 1; off < 32; off <<= 1) {
        s  += __shfl_xor(s,  off, 32);
        sq += __shfl_xor(sq, off, 32);
    }
    const float mean = s * (1.0f / 256.0f);
    const float var  = sq * (1.0f / 256.0f) - mean * mean;
    const float inv  = rsqrtf(var + 1e-5f);
#pragma unroll
    for (int j = 0; j < 8; ++j) {
        int c = lane * 8 + j;
        outH[row * 256 + c] = f2bf((vals[j] - mean) * inv * g[c] + bta[c]);
    }
}

// ---------------------------------------------------------------------------
// Generic bf16 WMMA GEMM: C[M,N] = epilogue(scale * A[M,K] @ B[K,N])
// B passed transposed: BT[N][K] bf16.  Block tile 128x64, KT=64, 8 waves.
// Double-buffered LDS: A tiles via per-lane async copies (ASYNCcnt),
// B (weight) tiles via Tensor Data Mover 2D descriptors (TENSORcnt).
// ---------------------------------------------------------------------------
__global__ __launch_bounds__(256) void gemm_bf16_kernel(
    const unsigned short* __restrict__ A, const unsigned short* __restrict__ BT,
    int M, int N, int K, float scale,
    const float* __restrict__ resid,
    float* __restrict__ outF, unsigned short* __restrict__ outH, int do_gelu)
{
    __shared__ __align__(16) unsigned short As[2][128 * 64]; // 2 x 16 KB
    __shared__ __align__(16) unsigned short Bs[2][64 * 64];  // 2 x  8 KB

    const int tid  = threadIdx.x;
    const int lane = tid & 31;
    const int wave = tid >> 5;
    const int wm = wave & 3;   // 4 M-strips of 32
    const int wn = wave >> 2;  // 2 N-strips of 32
    const int mBase = blockIdx.y * 128;
    const int nBase = blockIdx.x * 64;

    const int arow = tid >> 3;        // 0..31: this thread's base A row
    const int ac8  = (tid & 7) << 3;  // 16B column chunk

    // A tile: 4 async 16B copies per thread (rows arow, +32, +64, +96)
    auto issueA = [&](int buf, int kt) {
        const unsigned short* ag = A + (size_t)(mBase + arow) * K + kt + ac8;
        unsigned short* al = &As[buf][arow * 64 + ac8];
#pragma unroll
        for (int i = 0; i < 4; ++i) {
            async_cp16(al, ag);
            al += 32 * 64;
            ag += (size_t)32 * K;
        }
    };
    // B tile: one TDM 2D descriptor, issued by wave 0 only
    auto issueB = [&](int buf, int kt) {
        if (tid < 32)
            tdm_load_2d(&Bs[buf][0], BT + (size_t)nBase * K + kt,
                        /*tile_w=*/64, /*tile_h=*/64, /*stride=*/(unsigned)K);
    };

    v8f acc[2][2];
#pragma unroll
    for (int i = 0; i < 2; ++i)
#pragma unroll
        for (int j = 0; j < 2; ++j) acc[i][j] = vzero8();

    const int ntile = K >> 6;
    issueA(0, 0);
    issueB(0, 0);

    for (int t = 0; t < ntile; ++t) {
        const int cur = t & 1;
        if (t + 1 < ntile) {
            issueA(cur ^ 1, (t + 1) << 6);  // prefetch next tile
            issueB(cur ^ 1, (t + 1) << 6);
            wait_async4();                  // current A tile done (in-order)
            wait_tensor1();                 // current B tile done (in-order)
        } else {
            wait_async0();
            wait_tensor0();
        }
        __syncthreads();

        const unsigned short* Asb = As[cur];
        const unsigned short* Bsb = Bs[cur];
#pragma unroll
        for (int kk = 0; kk < 64; kk += 32) {
            v16bf a0 = lds_frag_a(Asb, wm * 32,      64, kk);
            v16bf a1 = lds_frag_a(Asb, wm * 32 + 16, 64, kk);
            v16bf b0 = lds_frag_b(Bsb, wn * 32,      64, kk);
            v16bf b1 = lds_frag_b(Bsb, wn * 32 + 16, 64, kk);
            acc[0][0] = wmma_bf16(a0, b0, acc[0][0]);
            acc[0][1] = wmma_bf16(a0, b1, acc[0][1]);
            acc[1][0] = wmma_bf16(a1, b0, acc[1][0]);
            acc[1][1] = wmma_bf16(a1, b1, acc[1][1]);
        }
        __syncthreads();
    }

    // Epilogue per ISA C/D layout: lane holds col lane%16, rows rb..rb+7
#pragma unroll
    for (int fm = 0; fm < 2; ++fm)
#pragma unroll
        for (int fn = 0; fn < 2; ++fn) {
            const int col = nBase + wn * 32 + fn * 16 + (lane & 15);
            const int rb  = mBase + wm * 32 + fm * 16 + ((lane >> 4) << 3);
            v8f a = acc[fm][fn];
#pragma unroll
            for (int r = 0; r < 8; ++r) {
                size_t idx = (size_t)(rb + r) * N + col;
                float v = a[r] * scale;
                if (resid)   v += resid[idx];
                if (do_gelu) v = 0.5f * v * (1.0f + erff(v * 0.70710678118654752f));
                if (outF)    outF[idx] = v;
                if (outH)    outH[idx] = f2bf(v);
            }
        }
}

// ---------------------------------------------------------------------------
// Fused attention: one block per (b,h). T=S=128, head dim 64, 8 waves.
// q/k tiles arrive via TDM 2D strided descriptors; scores kept in
// accumulator registers; softmax via intra-16-lane shuffles.
// ---------------------------------------------------------------------------
__global__ __launch_bounds__(256) void attention_kernel(
    const unsigned short* __restrict__ qk,  // [T,B, 2D] bf16 (pre-scaled by 1/8)
    const unsigned short* __restrict__ v,   // [T,B, D]  bf16 (pre-scaled by 1/8)
    unsigned short* __restrict__ attnout,   // [T,B, D]  bf16
    float* __restrict__ rawOut,             // [B,T,S] f32 (accumulated, pre-zeroed)
    float* __restrict__ wOut,               // [B,T,S] f32 (accumulated, pre-zeroed)
    int Bn, int NHn)
{
    __shared__ __align__(16) unsigned short smem[128 * 128]; // 32 KB: q|k, reused as p
    __shared__ __align__(16) unsigned short vTs[64 * 128];   // 16 KB: v transposed
    unsigned short* qs = smem;
    unsigned short* ks = smem + 128 * 64;
    unsigned short* pb = smem;

    const int tid  = threadIdx.x;
    const int lane = tid & 31;
    const int wave = tid >> 5;
    const int b = blockIdx.x / NHn;
    const int h = blockIdx.x % NHn;

    // TDM strided 2D loads of q, k [128 rows x 64 cols, row stride Bn*512]
    if (tid < 32) {
        const unsigned short* qbase = qk + (size_t)b * 512 + h * 128;
        tdm_load_2d(qs, qbase,      /*tile_w=*/64, /*tile_h=*/128,
                    /*stride=*/(unsigned)(Bn * 512));
        tdm_load_2d(ks, qbase + 64, /*tile_w=*/64, /*tile_h=*/128,
                    /*stride=*/(unsigned)(Bn * 512));
    }
    // Load v transposed (scatter; VGPR path): vTs[c][t]
    for (int i = tid; i < 8192; i += 256) {
        int t = i >> 6, c = i & 63;
        vTs[c * 128 + t] = v[((size_t)t * Bn + b) * 256 + h * 64 + c];
    }
    wait_tensor0();
    __syncthreads();

    // scores = q @ k^T : wave handles rows [wave*16, +16), all 128 cols in regs
    v8f acc[8];
#pragma unroll
    for (int n = 0; n < 8; ++n) acc[n] = vzero8();
#pragma unroll
    for (int kk = 0; kk < 64; kk += 32) {
        v16bf a = lds_frag_a(qs, wave * 16, 64, kk);
#pragma unroll
        for (int n = 0; n < 8; ++n) {
            v16bf bf = lds_frag_b(ks, n * 16, 64, kk);
            acc[n] = wmma_bf16(a, bf, acc[n]);
        }
    }
    __syncthreads();  // all q/k reads complete before pb aliasing writes

    const int rb = wave * 16 + ((lane >> 4) << 3);
    const int cl = lane & 15;
    const size_t obase = (size_t)b * (128 * 128);

    // attn_raw mean over heads
#pragma unroll
    for (int n = 0; n < 8; ++n)
#pragma unroll
        for (int r = 0; r < 8; ++r)
            atomicAdd(&rawOut[obase + (size_t)(rb + r) * 128 + n * 16 + cl],
                      acc[n][r] * 0.25f);

    // Softmax per row (row spread over 16 lanes x 8 frags)
#pragma unroll
    for (int r = 0; r < 8; ++r) {
        float m = -3.4e38f;
#pragma unroll
        for (int n = 0; n < 8; ++n) m = fmaxf(m, acc[n][r]);
#pragma unroll
        for (int off = 1; off < 16; off <<= 1) m = fmaxf(m, __shfl_xor(m, off, 32));
        float e[8], s = 0.0f;
#pragma unroll
        for (int n = 0; n < 8; ++n) { e[n] = expf(acc[n][r] - m); s += e[n]; }
#pragma unroll
        for (int off = 1; off < 16; off <<= 1) s += __shfl_xor(s, off, 32);
        const float inv = 1.0f / s;
#pragma unroll
        for (int n = 0; n < 8; ++n) {
            float pv = e[n] * inv;
            pb[(rb + r) * 128 + n * 16 + cl] = f2bf(pv);
            atomicAdd(&wOut[obase + (size_t)(rb + r) * 128 + n * 16 + cl], pv * 0.25f);
        }
    }
    __syncthreads();

    // out = p @ v : wave rows [wave*16,+16), 64 cols -> 4 frags, K=128
    v8f oacc[4];
#pragma unroll
    for (int n = 0; n < 4; ++n) oacc[n] = vzero8();
#pragma unroll
    for (int kk = 0; kk < 128; kk += 32) {
        v16bf a = lds_frag_a(pb, wave * 16, 128, kk);
#pragma unroll
        for (int n = 0; n < 4; ++n) {
            v16bf bf = lds_frag_b(vTs, n * 16, 128, kk);
            oacc[n] = wmma_bf16(a, bf, oacc[n]);
        }
    }
#pragma unroll
    for (int n = 0; n < 4; ++n)
#pragma unroll
        for (int r = 0; r < 8; ++r) {
            int t = rb + r;
            attnout[((size_t)t * Bn + b) * 256 + h * 64 + n * 16 + cl] = f2bf(oacc[n][r]);
        }
}

// ---------------------------------------------------------------------------
// Host orchestration
// ---------------------------------------------------------------------------
extern "C" void kernel_launch(void* const* d_in, const int* in_sizes, int n_in,
                              void* d_out, int out_size, void* d_ws, size_t ws_size,
                              hipStream_t stream) {
    (void)in_sizes; (void)n_in; (void)out_size; (void)ws_size;
    const int T_ = 128, Bn = 1024, C_ = 256, NH_ = 4;
    const size_t Mrows = (size_t)T_ * Bn;           // 131072
    const int Mi = (int)Mrows;

    const float* tok   = (const float*)d_in[0];
    const float* w_in  = (const float*)d_in[1];
    const float* ln1_g = (const float*)d_in[2];
    const float* ln1_b = (const float*)d_in[3];
    const float* w_qk  = (const float*)d_in[4];
    const float* w_v   = (const float*)d_in[5];
    const float* w_out = (const float*)d_in[6];
    const float* ln2_g = (const float*)d_in[7];
    const float* ln2_b = (const float*)d_in[8];
    const float* w_ff1 = (const float*)d_in[9];
    const float* w_ff2 = (const float*)d_in[10];
    const float* w_lin = (const float*)d_in[11];

    // Outputs: y [T,B,C], attn_weights [B,T,S], attn_raw [B,T,S]
    float* y   = (float*)d_out;
    float* aw  = y + (size_t)T_ * Bn * C_;          // +33554432
    float* raw = aw + (size_t)Bn * T_ * T_;         // +16777216

    // Workspace layout (with reuse)
    char* ws = (char*)d_ws;
    size_t off = 0;
    auto alloc = [&](size_t bytes) -> void* {
        void* p = ws + off;
        off += (bytes + 255) & ~(size_t)255;
        return p;
    };
    unsigned short* w_inT  = (unsigned short*)alloc((size_t)256 * 256 * 2);
    unsigned short* w_qkT  = (unsigned short*)alloc((size_t)512 * 256 * 2);
    unsigned short* w_vT   = (unsigned short*)alloc((size_t)256 * 256 * 2);
    unsigned short* w_outT = (unsigned short*)alloc((size_t)256 * 256 * 2);
    unsigned short* w_ff1T = (unsigned short*)alloc((size_t)512 * 256 * 2);
    unsigned short* w_ff2T = (unsigned short*)alloc((size_t)256 * 512 * 2);
    unsigned short* w_linT = (unsigned short*)alloc((size_t)256 * 256 * 2);
    unsigned short* bufA   = (unsigned short*)alloc(Mrows * 256 * 2); // tok_bf16 -> attnout
    float*          x_f32  = (float*)alloc(Mrows * 256 * 4);
    unsigned short* x_bf16 = (unsigned short*)alloc(Mrows * 256 * 2);
    unsigned short* bufN   = (unsigned short*)alloc(Mrows * 256 * 2); // xn -> x2n
    unsigned short* bufE   = (unsigned short*)alloc(Mrows * 512 * 2); // qk -> ffh
    unsigned short* bufF   = (unsigned short*)alloc(Mrows * 256 * 2); // v  -> x3
    float*          x2_f32 = (float*)alloc(Mrows * 256 * 4);

    const dim3 blk(256);

    // 0) zero the atomically-accumulated outputs (attn_weights + attn_raw, contiguous)
    {
        size_t n = 2ull * Bn * T_ * T_;
        zero_f32_kernel<<<dim3((unsigned)((n + 255) / 256)), blk, 0, stream>>>(aw, n);
    }
    // 1) tok -> bf16
    {
        size_t n = Mrows * C_;
        cvt_bf16_kernel<<<dim3((unsigned)((n + 255) / 256)), blk, 0, stream>>>(tok, bufA, n);
    }
    // 2) transpose+convert all weights
    wtrans_kernel<<<dim3((256 * 256 + 255) / 256), blk, 0, stream>>>(w_in,  w_inT,  256, 256);
    wtrans_kernel<<<dim3((256 * 512 + 255) / 256), blk, 0, stream>>>(w_qk,  w_qkT,  256, 512);
    wtrans_kernel<<<dim3((256 * 256 + 255) / 256), blk, 0, stream>>>(w_v,   w_vT,   256, 256);
    wtrans_kernel<<<dim3((256 * 256 + 255) / 256), blk, 0, stream>>>(w_out, w_outT, 256, 256);
    wtrans_kernel<<<dim3((256 * 512 + 255) / 256), blk, 0, stream>>>(w_ff1, w_ff1T, 256, 512);
    wtrans_kernel<<<dim3((512 * 256 + 255) / 256), blk, 0, stream>>>(w_ff2, w_ff2T, 512, 256);
    wtrans_kernel<<<dim3((256 * 256 + 255) / 256), blk, 0, stream>>>(w_lin, w_linT, 256, 256);

    const dim3 g256(256 / 64, (unsigned)(Mrows / 128)); // N=256
    const dim3 g512(512 / 64, (unsigned)(Mrows / 128)); // N=512

    // 3) x = tok @ w_in  (f32 + bf16 copies)
    gemm_bf16_kernel<<<g256, blk, 0, stream>>>(bufA, w_inT, Mi, 256, 256, 1.0f,
                                               nullptr, x_f32, x_bf16, 0);
    // 4) xn = LN1(x)
    layernorm_kernel<<<dim3((unsigned)(Mrows / 8)), blk, 0, stream>>>(x_f32, ln1_g, ln1_b, bufN);
    // 5) qk = (xn @ w_qk) / 8
    gemm_bf16_kernel<<<g512, blk, 0, stream>>>(bufN, w_qkT, Mi, 512, 256, 0.125f,
                                               nullptr, nullptr, bufE, 0);
    // 6) v = (x @ w_v) / 8   (un-normalized x, as in reference)
    gemm_bf16_kernel<<<g256, blk, 0, stream>>>(x_bf16, w_vT, Mi, 256, 256, 0.125f,
                                               nullptr, nullptr, bufF, 0);
    // 7) attention (writes attnout into bufA; atomically accumulates raw/weights means)
    attention_kernel<<<dim3((unsigned)(Bn * NH_)), blk, 0, stream>>>(bufE, bufF, bufA,
                                                                    raw, aw, Bn, NH_);
    // 8) x2 = attnout @ w_out + x
    gemm_bf16_kernel<<<g256, blk, 0, stream>>>(bufA, w_outT, Mi, 256, 256, 1.0f,
                                               x_f32, x2_f32, nullptr, 0);
    // 9) x2n = LN2(x2)
    layernorm_kernel<<<dim3((unsigned)(Mrows / 8)), blk, 0, stream>>>(x2_f32, ln2_g, ln2_b, bufN);
    // 10) ffh = gelu(x2n @ w_ff1)
    gemm_bf16_kernel<<<g512, blk, 0, stream>>>(bufN, w_ff1T, Mi, 512, 256, 1.0f,
                                               nullptr, nullptr, bufE, 1);
    // 11) x3 = ffh @ w_ff2 + x2
    gemm_bf16_kernel<<<g256, blk, 0, stream>>>(bufE, w_ff2T, Mi, 256, 512, 1.0f,
                                               x2_f32, nullptr, bufF, 0);
    // 12) y = x3 @ w_lin_out
    gemm_bf16_kernel<<<g256, blk, 0, stream>>>(bufF, w_linT, Mi, 256, 256, 1.0f,
                                               nullptr, y, nullptr, 0);
}
